// SpatialSelfAttention_42640435314778
// MI455X (gfx1250) — matrix operand order repaired and checked by reading, not compiled, not run on hardware
//
#include <hip/hip_runtime.h>

typedef _Float16 half16 __attribute__((ext_vector_type(16)));
typedef _Float16 half8  __attribute__((ext_vector_type(8)));
typedef float    f32x8  __attribute__((ext_vector_type(8)));

#define WMMA_F16(a, b, c) \
  __builtin_amdgcn_wmma_f32_16x16x32_f16(false, (a), false, (b), (short)0, (c), false, false)

constexpr int B_  = 8;
constexpr int C_  = 128;
constexpr int N_  = 4096;   // H*W
constexpr int KC_ = 32;

// Async global->LDS path (gfx1250 GLOBAL_LOAD_ASYNC_TO_LDS_B128, ASYNCcnt),
// guarded so the kernel still compiles if this toolchain lacks the builtins.
#if defined(__has_builtin)
#if __has_builtin(__builtin_amdgcn_global_load_async_to_lds_b128) && \
    __has_builtin(__builtin_amdgcn_s_wait_asynccnt)
#define USE_ASYNC_LDS 1
#endif
#endif

#if USE_ASYNC_LDS
#define ASYNC_WAIT(n) __builtin_amdgcn_s_wait_asynccnt(n)
// Builtin parameter types (from compiler diagnostic): GCC-vector int4 pointers
// qualified with addrspace(1) (source) and addrspace(3) (LDS dest).
typedef int i4v __attribute__((vector_size(16)));
typedef __attribute__((address_space(1))) i4v i4v_glob;
typedef __attribute__((address_space(3))) i4v i4v_lds;
static __device__ inline void cp16(const _Float16* g, _Float16* l) {
  // LDS generic address: low 32 bits are the LDS byte offset (aperture rule).
  __builtin_amdgcn_global_load_async_to_lds_b128(
      (i4v_glob*)(uintptr_t)g,
      (i4v_lds*)(uint32_t)(uintptr_t)l,
      0, 0);
}
#else
#define ASYNC_WAIT(n)
static __device__ inline void cp16(const _Float16* g, _Float16* l) {
  *(half8*)l = *(const half8*)g;
}
#endif

// ---------------------------------------------------------------------------
// Operand loaders for v_wmma_f32_16x16x32_f16 (wave32).
// A (16x32, MxK): lane l16 = row M; 16 contiguous halfs at
//   K = hf*16..+15.
// B (32x16, KxN): lane l16 = col N; 16 contiguous halfs at K = hf*16..+15.
// ---------------------------------------------------------------------------
static __device__ inline half16 ld_a(const _Float16* base, int strideHalfs,
                                     int l16, int hf) {
  union { half16 v; half8 h[2]; } u;
  const _Float16* p = base + (size_t)l16 * strideHalfs;
  u.h[0] = *(const half8*)(p + hf * 16);
  u.h[1] = *(const half8*)(p + hf * 16 + 8);
  return u.v;
}

static __device__ inline half16 ld_b(const _Float16* base, int strideHalfs,
                                     int l16, int hf) {
  union { half16 v; half8 h[2]; } u;
  const _Float16* p = base + (size_t)l16 * strideHalfs + hf * 16;
  u.h[0] = *(const half8*)(p);
  u.h[1] = *(const half8*)(p + 8);
  return u.v;
}

static __device__ inline float redmax16(float v) {
#pragma unroll
  for (int m = 1; m < 16; m <<= 1) v = fmaxf(v, __shfl_xor(v, m, 32));
  return v;
}
static __device__ inline float redsum16(float v) {
#pragma unroll
  for (int m = 1; m < 16; m <<= 1) v += __shfl_xor(v, m, 32);
  return v;
}

// ---------------------------------------------------------------------------
// Kernel 0: convert f32 weights to f16 copies in workspace.
// ---------------------------------------------------------------------------
__global__ void cvt_weights(const float* __restrict__ Wq, const float* __restrict__ Wk,
                            const float* __restrict__ Wv, const float* __restrict__ Wo,
                            _Float16* __restrict__ hq, _Float16* __restrict__ hk,
                            _Float16* __restrict__ hv, _Float16* __restrict__ ho) {
  int i = blockIdx.x * 256 + threadIdx.x;
  if (i < KC_ * C_) { hq[i] = (_Float16)Wq[i]; hk[i] = (_Float16)Wk[i]; }
  if (i < C_ * C_)  { hv[i] = (_Float16)Wv[i]; ho[i] = (_Float16)Wo[i]; }
}

// ---------------------------------------------------------------------------
// Kernel 1: QKV projection via WMMA (one wave = 16 spatial positions).
//   Qh, Kh : [B][N][32]   (row per position, f16)
//   Vt     : [B][C][N]    (channel-major, f16)
// ---------------------------------------------------------------------------
__global__ __launch_bounds__(32) void proj_qkv(
    const float* __restrict__ x,
    const _Float16* __restrict__ Whq, const _Float16* __restrict__ Whk,
    const _Float16* __restrict__ Whv,
    _Float16* __restrict__ Qh, _Float16* __restrict__ Kh,
    _Float16* __restrict__ Vt) {
  __shared__ _Float16 XS[16 * 136];  // [n][c], padded stride

  const int lane = threadIdx.x & 31;
  const int l16 = lane & 15, hf = (lane >> 4) & 1;
  const int b  = blockIdx.x / (N_ / 16);
  const int n0 = (blockIdx.x % (N_ / 16)) * 16;

  // Transpose x[b, c, n0:n0+16] (f32) -> XS[n][c] (f16)
#pragma unroll
  for (int i = 0; i < 4; ++i) {
    const int c = lane + i * 32;
    const float* src = x + ((size_t)b * C_ + c) * N_ + n0;
    float v[16];
#pragma unroll
    for (int j = 0; j < 4; ++j) {
      float4 f = ((const float4*)src)[j];
      v[4 * j + 0] = f.x; v[4 * j + 1] = f.y;
      v[4 * j + 2] = f.z; v[4 * j + 3] = f.w;
    }
#pragma unroll
    for (int j = 0; j < 16; ++j) XS[j * 136 + c] = (_Float16)v[j];
  }

  half16 ax[4];
#pragma unroll
  for (int cb = 0; cb < 4; ++cb) ax[cb] = ld_a(XS + cb * 32, 136, l16, hf);

#pragma unroll
  for (int ot = 0; ot < 2; ++ot) {
    f32x8 dq, dk;
#pragma unroll
    for (int r = 0; r < 8; ++r) { dq[r] = 0.0f; dk[r] = 0.0f; }
#pragma unroll
    for (int cb = 0; cb < 4; ++cb) {
      dq = WMMA_F16(ax[cb], ld_b(Whq + ot * 16 * C_ + cb * 32, C_, l16, hf), dq);
      dk = WMMA_F16(ax[cb], ld_b(Whk + ot * 16 * C_ + cb * 32, C_, l16, hf), dk);
    }
#pragma unroll
    for (int r = 0; r < 8; ++r) {
      const int n = n0 + 8 * hf + r;
      Qh[(size_t)(b * N_ + n) * KC_ + ot * 16 + l16] = (_Float16)dq[r];
      Kh[(size_t)(b * N_ + n) * KC_ + ot * 16 + l16] = (_Float16)dk[r];
    }
  }

#pragma unroll
  for (int ot = 0; ot < 8; ++ot) {
    f32x8 dv;
#pragma unroll
    for (int r = 0; r < 8; ++r) dv[r] = 0.0f;
#pragma unroll
    for (int cb = 0; cb < 4; ++cb)
      dv = WMMA_F16(ax[cb], ld_b(Whv + ot * 16 * C_ + cb * 32, C_, l16, hf), dv);
#pragma unroll
    for (int r = 0; r < 8; ++r) {
      const int n = n0 + 8 * hf + r;
      Vt[(size_t)b * C_ * N_ + (size_t)(ot * 16 + l16) * N_ + n] = (_Float16)dv[r];
    }
  }
}

// ---------------------------------------------------------------------------
// Kernel 2: flash attention. 4 waves/block, 32 queries per wave (two A-tiles,
// so every K/V B-tile read from LDS feeds two WMMAs); 64-key blocks,
// double-buffered K/V tiles staged with async global->LDS copies.
// Per key block per wave: 8 QK^T WMMAs + 32 PV WMMAs, online softmax (exp2).
// ---------------------------------------------------------------------------
constexpr int KB_   = 64;           // keys per block iteration
constexpr int NBLK_ = N_ / KB_;     // 64 iterations
constexpr int KSTR_ = 40;           // KT row stride (halfs), padded
constexpr int VSTR_ = 72;           // VS/PS row stride (halfs), padded
constexpr int QBLK_ = 128;          // queries per block (4 waves x 32)

// Stage K/V tile for key block jb into LDS buffers (10 x 16B per thread).
static __device__ inline void stage_tile(int tid, int b, int jb,
                                         const _Float16* Kh, const _Float16* Vt,
                                         _Float16* KTb, _Float16* VSb) {
#pragma unroll
  for (int i = 0; i < 2; ++i) {            // K: 64 rows x 32 halfs = 256 chunks
    const int id = tid + i * 128;
    const int row = id >> 2, ch = id & 3;
    cp16(Kh + (size_t)(b * N_ + jb * KB_ + row) * KC_ + ch * 8,
         KTb + row * KSTR_ + ch * 8);
  }
#pragma unroll
  for (int i = 0; i < 8; ++i) {            // V: 128 rows x 64 halfs = 1024 chunks
    const int id = tid + i * 128;
    const int row = id >> 3, ch = id & 7;
    cp16(Vt + (size_t)b * C_ * N_ + (size_t)row * N_ + jb * KB_ + ch * 8,
         VSb + row * VSTR_ + ch * 8);
  }
}

__global__ __launch_bounds__(128) void flash_attn(
    const _Float16* __restrict__ Qh, const _Float16* __restrict__ Kh,
    const _Float16* __restrict__ Vt, _Float16* __restrict__ O) {
  __shared__ _Float16 KT[2][KB_ * KSTR_];    // [key][kc]      2 x  5 KB
  __shared__ _Float16 VS[2][C_ * VSTR_];     // [c][key]       2 x 18 KB
  __shared__ _Float16 PS[4][32 * VSTR_];     // per-wave P     4 x 4.5 KB

  const int tid = threadIdx.x;
  const int wave = tid >> 5, lane = tid & 31;
  const int l16 = lane & 15, hf = (lane >> 4) & 1;
  const int b  = blockIdx.x / (N_ / QBLK_);
  const int n0 = (blockIdx.x % (N_ / QBLK_)) * QBLK_ + wave * 32;

  half16 aq[2];
#pragma unroll
  for (int qt = 0; qt < 2; ++qt)
    aq[qt] = ld_a(Qh + (size_t)(b * N_ + n0 + qt * 16) * KC_, KC_, l16, hf);

  f32x8 acc[2][8];
  float mrow[2][8], lrow[2][8];
#pragma unroll
  for (int qt = 0; qt < 2; ++qt)
#pragma unroll
    for (int r = 0; r < 8; ++r) {
      mrow[qt][r] = -1e30f; lrow[qt][r] = 0.0f;
#pragma unroll
      for (int ct = 0; ct < 8; ++ct) acc[qt][ct][r] = 0.0f;
    }

  const float scl2 = 0.25505858f;  // (1/sqrt(KC)) * log2(e); exp via exp2
  _Float16* Pw = PS[wave];

  stage_tile(tid, b, 0, Kh, Vt, KT[0], VS[0]);

  for (int jb = 0; jb < NBLK_; ++jb) {
    const int cur = jb & 1;
    if (jb + 1 < NBLK_) {
      stage_tile(tid, b, jb + 1, Kh, Vt, KT[cur ^ 1], VS[cur ^ 1]);
      ASYNC_WAIT(10);   // in-order: ensures current tile's 10 copies landed
    } else {
      ASYNC_WAIT(0);
    }
    __syncthreads();

    const _Float16* KTc = KT[cur];
    const _Float16* VSc = VS[cur];

    // --- K B-tiles loaded once, reused by both query tiles ---
    half16 bk[4];
#pragma unroll
    for (int si = 0; si < 4; ++si)
      bk[si] = ld_b(KTc + si * 16 * KSTR_, KSTR_, l16, hf);

    f32x8 z;
#pragma unroll
    for (int r = 0; r < 8; ++r) z[r] = 0.0f;

#pragma unroll
    for (int qt = 0; qt < 2; ++qt) {
      // --- S = Q * K^T over 64 keys: 4 WMMAs ---
      f32x8 s[4];
#pragma unroll
      for (int si = 0; si < 4; ++si) s[si] = WMMA_F16(aq[qt], bk[si], z);

      // --- online softmax over 64 keys at once ---
#pragma unroll
      for (int r = 0; r < 8; ++r) {
        float xv[4];
#pragma unroll
        for (int si = 0; si < 4; ++si) xv[si] = s[si][r] * scl2;
        float mx = fmaxf(fmaxf(xv[0], xv[1]), fmaxf(xv[2], xv[3]));
        mx = redmax16(mx);
        const float mnew = fmaxf(mrow[qt][r], mx);
        const float corr = exp2f(mrow[qt][r] - mnew);
        float p[4], psum = 0.0f;
#pragma unroll
        for (int si = 0; si < 4; ++si) { p[si] = exp2f(xv[si] - mnew); psum += p[si]; }
        lrow[qt][r] = lrow[qt][r] * corr + redsum16(psum);
        mrow[qt][r] = mnew;
#pragma unroll
        for (int ct = 0; ct < 8; ++ct) acc[qt][ct][r] *= corr;
        const int qr = qt * 16 + 8 * hf + r;
#pragma unroll
        for (int si = 0; si < 4; ++si)
          Pw[qr * VSTR_ + si * 16 + l16] = (_Float16)p[si];
      }
    }

    // --- acc += P * V: each V B-tile feeds both query tiles (32 WMMAs) ---
    half16 ap[2][2];
#pragma unroll
    for (int qt = 0; qt < 2; ++qt) {
      ap[qt][0] = ld_a(Pw + qt * 16 * VSTR_, VSTR_, l16, hf);
      ap[qt][1] = ld_a(Pw + qt * 16 * VSTR_ + 32, VSTR_, l16, hf);
    }
#pragma unroll
    for (int ct = 0; ct < 8; ++ct) {
      const half16 bv0 = ld_b(VSc + ct * 16 * VSTR_, VSTR_, l16, hf);
      const half16 bv1 = ld_b(VSc + ct * 16 * VSTR_ + 32, VSTR_, l16, hf);
      acc[0][ct] = WMMA_F16(ap[0][0], bv0, acc[0][ct]);
      acc[0][ct] = WMMA_F16(ap[0][1], bv1, acc[0][ct]);
      acc[1][ct] = WMMA_F16(ap[1][0], bv0, acc[1][ct]);
      acc[1][ct] = WMMA_F16(ap[1][1], bv1, acc[1][ct]);
    }
    __syncthreads();
  }

  // --- epilogue: normalize and store O[b][n][c] f16 ---
#pragma unroll
  for (int qt = 0; qt < 2; ++qt)
#pragma unroll
    for (int r = 0; r < 8; ++r) {
      const float inv = 1.0f / lrow[qt][r];
      const int n = n0 + qt * 16 + 8 * hf + r;
      _Float16* orow = O + (size_t)(b * N_ + n) * C_;
#pragma unroll
      for (int ct = 0; ct < 8; ++ct)
        orow[ct * 16 + l16] = (_Float16)(acc[qt][ct][r] * inv);
    }
}

// ---------------------------------------------------------------------------
// Kernel 3: output projection + residual.  y = x + gamma * (Wo @ attn_out)
// ---------------------------------------------------------------------------
__global__ __launch_bounds__(32) void out_proj(
    const float* __restrict__ x, const _Float16* __restrict__ O,
    const _Float16* __restrict__ Who, const float* __restrict__ gamma,
    float* __restrict__ y) {
  const int lane = threadIdx.x & 31;
  const int l16 = lane & 15, hf = (lane >> 4) & 1;
  const int b  = blockIdx.x / (N_ / 16);
  const int n0 = (blockIdx.x % (N_ / 16)) * 16;
  const float g = gamma[0];

  half16 ao[4];
#pragma unroll
  for (int cb = 0; cb < 4; ++cb)
    ao[cb] = ld_a(O + (size_t)(b * N_ + n0) * C_ + cb * 32, C_, l16, hf);

#pragma unroll
  for (int ot = 0; ot < 8; ++ot) {
    f32x8 d;
#pragma unroll
    for (int r = 0; r < 8; ++r) d[r] = 0.0f;
#pragma unroll
    for (int cb = 0; cb < 4; ++cb)
      d = WMMA_F16(ao[cb], ld_b(Who + ot * 16 * C_ + cb * 32, C_, l16, hf), d);
#pragma unroll
    for (int r = 0; r < 8; ++r) {
      const int n = n0 + 8 * hf + r;
      const size_t idx = ((size_t)b * C_ + ot * 16 + l16) * N_ + n;
      y[idx] = x[idx] + g * d[r];
    }
  }
}

// ---------------------------------------------------------------------------
extern "C" void kernel_launch(void* const* d_in, const int* in_sizes, int n_in,
                              void* d_out, int out_size, void* d_ws, size_t ws_size,
                              hipStream_t stream) {
  const float* x  = (const float*)d_in[0];
  const float* Wq = (const float*)d_in[1];
  const float* Wk = (const float*)d_in[2];
  const float* Wv = (const float*)d_in[3];
  const float* Wo = (const float*)d_in[4];
  const float* gm = (const float*)d_in[5];
  float* y = (float*)d_out;

  char* ws = (char*)d_ws;
  _Float16* Whq = (_Float16*)(ws + 0);            //  8 KB
  _Float16* Whk = (_Float16*)(ws + 8192);         //  8 KB
  _Float16* Whv = (_Float16*)(ws + 16384);        // 32 KB
  _Float16* Who = (_Float16*)(ws + 49152);        // 32 KB
  _Float16* Qh  = (_Float16*)(ws + 81920);                          // 2 MB
  _Float16* Kh  = (_Float16*)(ws + 81920 + 2097152);                // 2 MB
  _Float16* Vt  = (_Float16*)(ws + 81920 + 2 * 2097152);            // 8 MB
  _Float16* O   = (_Float16*)(ws + 81920 + 2 * 2097152 + 8388608);  // 8 MB

  cvt_weights<<<64, 256, 0, stream>>>(Wq, Wk, Wv, Wo, Whq, Whk, Whv, Who);
  proj_qkv<<<B_ * (N_ / 16), 32, 0, stream>>>(x, Whq, Whk, Whv, Qh, Kh, Vt);
  flash_attn<<<B_ * (N_ / QBLK_), 128, 0, stream>>>(Qh, Kh, Vt, O);
  out_proj<<<B_ * (N_ / 16), 32, 0, stream>>>(x, O, Who, gm, y);
}
